// multihead_self_attention_48773648613974
// MI455X (gfx1250) — compile-verified, matrix operand order checked by reading, not run on hardware
//
#include <hip/hip_runtime.h>
#include <hip/hip_bf16.h>

// ---------------------------------------------------------------------------
// Multi-head causal self-attention with RoPE, bf16 WMMA pipeline for gfx1250.
// B=4, S=2048, D=1024, H=16, dk=64.
// CDNA5 paths: v_wmma_f32_16x16x32_bf16, global_load_async_to_lds_b128
// (ASYNCcnt double-buffering), tensor_load_to_lds (TDM + TENSORcnt),
// ds_swizzle_b32 reductions.
// ---------------------------------------------------------------------------

typedef __bf16 bf16;
typedef __attribute__((ext_vector_type(16))) __bf16 v16bf;
typedef __attribute__((ext_vector_type(8)))  float  v8f;
typedef __attribute__((ext_vector_type(4)))  int    v4i_s;
typedef __attribute__((ext_vector_type(8)))  int    v8i_s;

union FragU {
    v16bf v;
    uint4 q[2];
};
union Cv4 {           // 4 packed bf16 = 8 bytes
    __bf16 h[4];
    uint2 u;
};

#define WMMA_BF16(a, b, c) \
    __builtin_amdgcn_wmma_f32_16x16x32_bf16(false, (a), false, (b), (short)0, (c), false, false)

// ds_swizzle XOR within 16-lane halves (and=0x1f, or=0, xor in [14:10])
#define SWZ_XOR(x, imm) \
    __int_as_float(__builtin_amdgcn_ds_swizzle(__float_as_int(x), (imm)))
#define SWZ_X8  0x201F
#define SWZ_X4  0x101F
#define SWZ_X2  0x081F
#define SWZ_X1  0x041F

// Async global->LDS copy of 16 bytes per lane (tracked by ASYNCcnt).
__device__ __forceinline__ void async_copy_b128(unsigned lds_byte_addr,
                                                const void* gaddr)
{
    unsigned long long ga = (unsigned long long)gaddr;
    asm volatile("global_load_async_to_lds_b128 %0, %1, off"
                 :: "v"(lds_byte_addr), "v"(ga) : "memory");
}
__device__ __forceinline__ void wait_async(int n)
{
    if (n == 0) asm volatile("s_wait_asynccnt 0" ::: "memory");
    else        asm volatile("s_wait_asynccnt 8" ::: "memory");
}
__device__ __forceinline__ unsigned lds_addr_of(const void* p)
{
    return (unsigned)(unsigned long long)p;   // LDS aperture: offset in [31:0]
}

// ---------------------------------------------------------------------------
// Tensor Data Mover: 2D bf16 tile load, global -> LDS, with LDS padding.
// D# per CDNA5 ISA ch.8: group0 (4 SGPRs) + group1 (8 SGPRs); 2D => groups
// 2/3 are NULL. Issued by one wave; completion via s_wait_tensorcnt.
// ---------------------------------------------------------------------------
__device__ __forceinline__ void tdm_load_2d_bf16(
    unsigned lds_byte_addr, const void* gaddr,
    int tile_w_elems, int tile_h_rows,
    int tensor_w_elems, int tensor_h_rows, int row_stride_elems,
    int pad_interval_code, int pad_amount_code, int pad_enable)
{
    unsigned long long ga = (unsigned long long)gaddr;
    v4i_s g0;
    g0.x = 1;                                        // count=1, user descriptor
    g0.y = (int)lds_byte_addr;                       // lds_addr [63:32]
    g0.z = (int)(unsigned)(ga & 0xFFFFFFFFull);      // global_addr [95:64]
    g0.w = (int)((unsigned)((ga >> 32) & 0x1FFFFFFull) | (2u << 30)); // +type=2
    v8i_s g1;
    g1.s0 = (1 << 16)                                // data_size=1 (2 bytes)
          | (pad_enable << 20)
          | ((pad_interval_code & 7) << 22)
          | ((pad_amount_code & 0x7F) << 25);
    g1.s1 = (tensor_w_elems & 0xFFFF) << 16;         // tensor_dim0[15:0]
    g1.s2 = ((unsigned)tensor_w_elems >> 16)         // tensor_dim0[31:16]
          | ((tensor_h_rows & 0xFFFF) << 16);        // tensor_dim1[15:0]
    g1.s3 = ((unsigned)tensor_h_rows >> 16)          // tensor_dim1[31:16]
          | ((tile_w_elems & 0xFFFF) << 16);         // tile_dim0
    g1.s4 = (tile_h_rows & 0xFFFF);                  // tile_dim1 (tile_dim2=0)
    g1.s5 = row_stride_elems;                        // tensor_dim0_stride[31:0]
    g1.s6 = 0;                                       // stride hi / dim1_stride lo
    g1.s7 = 0;                                       // dim1_stride hi
    asm volatile("tensor_load_to_lds %0, %1"
                 :: "s"(g0), "s"(g1) : "memory");
}

static constexpr int Bsz = 4;
static constexpr int S   = 2048;
static constexpr int D   = 1024;
static constexpr int H   = 16;
static constexpr int DK  = 64;
static constexpr int LDS_STRIDE = 40; // 32 + 8 pad (80 bytes/row, 16B aligned)
static constexpr int KST_STRIDE = 72; // 64 + 8 pad (144 bytes/row, 16B aligned)

// ---------------------------------------------------------------------------
// GEMM: C[M=8192, N=1024] = A[M,1024] * W^T   (y[m,n] = sum_k A[m,k]*W[n,k])
// Block tile 128x128, K-step 32. 256 threads = 8 waves; wave tile 64x32.
// A_BF16 path loads its A tile with one TDM descriptor per k-step.
// ---------------------------------------------------------------------------
template<bool A_BF16, bool OUT_F32>
__global__ __launch_bounds__(256)
void gemm_wmma(const void* __restrict__ Ap, const float* __restrict__ W,
               void* __restrict__ Cp)
{
    __shared__ __bf16 As[128 * LDS_STRIDE];
    __shared__ __bf16 Bs[128 * LDS_STRIDE];

    const int t    = threadIdx.x;
    const int m0   = blockIdx.x * 128;
    const int n0   = blockIdx.y * 128;
    const int wave = t >> 5;
    const int lane = t & 31;
    const int ln   = lane & 15;
    const int lh   = lane >> 4;
    const int wm   = (wave & 1) * 64;   // wave row offset in block tile
    const int wn   = (wave >> 1) * 32;  // wave col offset in block tile

    v8f acc[8];
    const v8f zero8 = {0.f,0.f,0.f,0.f,0.f,0.f,0.f,0.f};
#pragma unroll
    for (int i = 0; i < 8; ++i) acc[i] = zero8;

    for (int k0 = 0; k0 < D; k0 += 32) {
        // ---- load A tile -> LDS (bf16) ----
        if (A_BF16) {
            // One TDM 2D descriptor moves the whole 128x32 bf16 tile into LDS
            // with hardware row padding (16 DW data + 4 DW pad = stride 40).
            const __bf16* Ab = reinterpret_cast<const __bf16*>(Ap);
            if (wave == 0) {
                tdm_load_2d_bf16(lds_addr_of(As),
                                 Ab + (size_t)m0 * D + k0,
                                 /*tile_w=*/32, /*tile_h=*/128,
                                 /*tensor_w=*/D, /*tensor_h=*/8192,
                                 /*row_stride=*/D,
                                 /*pad_interval: 16 DW=*/3,
                                 /*pad_amount: 4 DW=*/3,
                                 /*pad_enable=*/1);
            }
        } else {
            const float* Af = reinterpret_cast<const float*>(Ap);
            const int lr = t >> 3;          // 0..31
            const int lc = (t & 7) * 4;     // 0,4,...,28
#pragma unroll
            for (int it = 0; it < 4; ++it) {
                const int row = it * 32 + lr;
                float4 f = *(const float4*)(Af + (size_t)(m0 + row) * D + k0 + lc);
                Cv4 cv;
                cv.h[0] = (__bf16)f.x; cv.h[1] = (__bf16)f.y;
                cv.h[2] = (__bf16)f.z; cv.h[3] = (__bf16)f.w;
                *(uint2*)&As[row * LDS_STRIDE + lc] = cv.u;
            }
            if (k0 + 32 < D)
                __builtin_prefetch(Af + (size_t)(m0 + lr) * D + k0 + 32 + lc, 0, 2);
        }
        // ---- load W tile (rows n, contiguous k) -> LDS (bf16) ----
        {
            const int lr = t >> 3;
            const int lc = (t & 7) * 4;
#pragma unroll
            for (int it = 0; it < 4; ++it) {
                const int row = it * 32 + lr;
                float4 f = *(const float4*)(W + (size_t)(n0 + row) * D + k0 + lc);
                Cv4 cv;
                cv.h[0] = (__bf16)f.x; cv.h[1] = (__bf16)f.y;
                cv.h[2] = (__bf16)f.z; cv.h[3] = (__bf16)f.w;
                *(uint2*)&Bs[row * LDS_STRIDE + lc] = cv.u;
            }
            if (k0 + 32 < D)
                __builtin_prefetch(W + (size_t)(n0 + lr) * D + k0 + 32 + lc, 0, 2);
        }
        if (A_BF16 && wave == 0)
            __builtin_amdgcn_s_wait_tensorcnt(0);   // tile landed in LDS
        __syncthreads();

        // ---- fragments + WMMA ----
        v16bf af[4];
#pragma unroll
        for (int i = 0; i < 4; ++i) {
            const __bf16* p = &As[(wm + i * 16 + ln) * LDS_STRIDE];
            FragU f;
            f.q[0] = *(const uint4*)(p + lh * 8);
            f.q[1] = *(const uint4*)(p + 16 + lh * 8);
            af[i] = f.v;
        }
        v16bf bfr[2];
#pragma unroll
        for (int j = 0; j < 2; ++j) {
            const __bf16* p = &Bs[(wn + j * 16 + ln) * LDS_STRIDE];
            FragU f;
            f.q[0] = *(const uint4*)(p + lh * 8);
            f.q[1] = *(const uint4*)(p + 16 + lh * 8);
            bfr[j] = f.v;
        }
#pragma unroll
        for (int i = 0; i < 4; ++i)
#pragma unroll
            for (int j = 0; j < 2; ++j)
                acc[i * 2 + j] = WMMA_BF16(af[i], bfr[j], acc[i * 2 + j]);

        __syncthreads();
    }

    // ---- epilogue ----
#pragma unroll
    for (int i = 0; i < 4; ++i) {
#pragma unroll
        for (int j = 0; j < 2; ++j) {
#pragma unroll
            for (int r = 0; r < 8; ++r) {
                const int row = m0 + wm + i * 16 + r + 8 * lh;
                const int col = n0 + wn + j * 16 + ln;
                const float val = acc[i * 2 + j][r];
                if (OUT_F32) {
                    reinterpret_cast<float*>(Cp)[(size_t)row * D + col] = val;
                } else {
                    reinterpret_cast<__bf16*>(Cp)[(size_t)row * D + col] = (__bf16)val;
                }
            }
        }
    }
}

// ---------------------------------------------------------------------------
// RoPE + head-major pack: y[B,S,H*dk] (bf16) ->
//   q,k: out[B,H,S,dk] with RoPE;  v: out[B,H,dk,S] transposed (no rope)
// ---------------------------------------------------------------------------
__global__ __launch_bounds__(256)
void rope_pack(const unsigned short* __restrict__ yraw,
               unsigned short* __restrict__ outraw,
               int do_rope, int transpose_v)
{
    const __bf16* y = reinterpret_cast<const __bf16*>(yraw);
    __bf16* out     = reinterpret_cast<__bf16*>(outraw);
    const int bs = blockIdx.x;          // 0..B*S-1
    const int b  = bs >> 11;
    const int s  = bs & (S - 1);
    const __bf16* row = y + (size_t)bs * D;

    for (int p = threadIdx.x; p < D / 2; p += 256) {
        const int h = p >> 5;           // head (32 pairs per head)
        const int i = p & 31;           // pair index within head
        const float x1 = (float)row[h * DK + 2 * i];
        const float x2 = (float)row[h * DK + 2 * i + 1];
        float r1, r2;
        if (do_rope) {
            const float freq = __powf(10000.f, -(float)i * (1.0f / 32.0f));
            const float ang  = (float)s * freq;
            float sn, cs;
            __sincosf(ang, &sn, &cs);
            r1 = cs * x1 - sn * x2;
            r2 = sn * x1 + cs * x2;
        } else {
            r1 = x1; r2 = x2;
        }
        if (transpose_v) {
            __bf16* dst = out + ((size_t)(b * H + h) * DK) * S;   // [b,h,j,s]
            dst[(size_t)(2 * i) * S + s]     = (__bf16)r1;
            dst[(size_t)(2 * i + 1) * S + s] = (__bf16)r2;
        } else {
            __bf16* dst = out + (((size_t)(b * H + h) * S + s) * DK) + 2 * i;
            dst[0] = (__bf16)r1;
            dst[1] = (__bf16)r2;
        }
    }
}

// ---------------------------------------------------------------------------
// Flash attention (causal), one wave per 16 query rows of one (b,h).
// q,k: [B,H,S,dk] bf16; v: [B,H,dk,S] bf16 (pre-transposed).
// K tiles staged to LDS via global_load_async_to_lds_b128 (double-buffered).
// grid = B*H*(S/16) = 8192 blocks of 32 threads.
// ---------------------------------------------------------------------------
__global__ __launch_bounds__(32)
void flash_attn(const unsigned short* __restrict__ qraw,
                const unsigned short* __restrict__ kraw,
                const unsigned short* __restrict__ vraw,
                unsigned short* __restrict__ attnraw)
{
    __shared__ __bf16 Pst[16 * LDS_STRIDE];     // P tile 16x32
    __shared__ __bf16 Kst[2][32 * KST_STRIDE];  // K tile 32x64, double-buffered

    const int blk = blockIdx.x;
    const int qt  = blk & 127;
    const int h   = (blk >> 7) & 15;
    const int b   = blk >> 11;
    const int q0  = qt * 16;

    const int lane = threadIdx.x;
    const int ln   = lane & 15;
    const int lh   = lane >> 4;

    const size_t headoff = ((size_t)(b * H + h)) * S * DK;
    const __bf16* qb  = reinterpret_cast<const __bf16*>(qraw) + headoff;
    const __bf16* kb  = reinterpret_cast<const __bf16*>(kraw) + headoff;
    const __bf16* vbT = reinterpret_cast<const __bf16*>(vraw) + headoff; // [dk][S]
    __bf16* attn      = reinterpret_cast<__bf16*>(attnraw);

    // Q fragments (16 rows x 64 d), A-layout, held in registers.
    v16bf aq[2];
#pragma unroll
    for (int f = 0; f < 2; ++f) {
        const __bf16* p = qb + (size_t)(q0 + ln) * DK + f * 32;
        FragU fr;
        fr.q[0] = *(const uint4*)(p + lh * 8);
        fr.q[1] = *(const uint4*)(p + 16 + lh * 8);
        aq[f] = fr.v;
    }

    const v8f zero8 = {0.f,0.f,0.f,0.f,0.f,0.f,0.f,0.f};
    v8f oacc[4];
#pragma unroll
    for (int i = 0; i < 4; ++i) oacc[i] = zero8;
    float mrow[8], lrow[8];
#pragma unroll
    for (int r = 0; r < 8; ++r) { mrow[r] = -1e30f; lrow[r] = 0.f; }

    const float scale = 0.125f;   // 1/sqrt(64)
    const int kv_end = q0 + 15;

    // Issue one 32x64 K tile into Kst[buf] (8 async b128 ops per wave).
    auto issue_k = [&](int kv, int buf) {
#pragma unroll
        for (int i = 0; i < 8; ++i) {
            const int c   = i * 32 + lane;  // chunk 0..255
            const int row = c >> 3;         // 0..31
            const int col = (c & 7) * 8;    // bf16 elems 0..56
            async_copy_b128(lds_addr_of(&Kst[buf][row * KST_STRIDE + col]),
                            kb + (size_t)(kv + row) * DK + col);
        }
    };

    int buf = 0;
    issue_k(0, 0);

    for (int kv = 0; kv <= kv_end; kv += 32) {
        const bool have_next = (kv + 32 <= kv_end);
        if (have_next) issue_k(kv + 32, buf ^ 1);
        wait_async(have_next ? 8 : 0);   // in-order: current tile landed

        // ---- K fragments (B-layout) from LDS ----
        v16bf bk[2][2];
#pragma unroll
        for (int nt = 0; nt < 2; ++nt) {
            const __bf16* kr = &Kst[buf][(nt * 16 + ln) * KST_STRIDE];
#pragma unroll
            for (int dh = 0; dh < 2; ++dh) {
                FragU fr;
                fr.q[0] = *(const uint4*)(kr + dh * 32 + lh * 8);
                fr.q[1] = *(const uint4*)(kr + dh * 32 + 16 + lh * 8);
                bk[dh][nt] = fr.v;
            }
        }

        v8f s0 = WMMA_BF16(aq[0], bk[0][0], zero8);
        s0     = WMMA_BF16(aq[1], bk[1][0], s0);
        v8f s1 = WMMA_BF16(aq[0], bk[0][1], zero8);
        s1     = WMMA_BF16(aq[1], bk[1][1], s1);

        // ---- online softmax (f32), fully unrolled ----
#pragma unroll
        for (int r = 0; r < 8; ++r) {
            const int row = q0 + r + 8 * lh;
            float a0 = s0[r] * scale;
            float a1 = s1[r] * scale;
            if (kv + ln > row)      a0 = -1e30f;
            if (kv + 16 + ln > row) a1 = -1e30f;

            float tm = fmaxf(a0, a1);
            tm = fmaxf(tm, SWZ_XOR(tm, SWZ_X8));
            tm = fmaxf(tm, SWZ_XOR(tm, SWZ_X4));
            tm = fmaxf(tm, SWZ_XOR(tm, SWZ_X2));
            tm = fmaxf(tm, SWZ_XOR(tm, SWZ_X1));

            const float mnew = fmaxf(mrow[r], tm);
            const float corr = __expf(mrow[r] - mnew);
            const float e0 = __expf(a0 - mnew);
            const float e1 = __expf(a1 - mnew);

            float ps = e0 + e1;
            ps += SWZ_XOR(ps, SWZ_X8);
            ps += SWZ_XOR(ps, SWZ_X4);
            ps += SWZ_XOR(ps, SWZ_X2);
            ps += SWZ_XOR(ps, SWZ_X1);

            lrow[r] = lrow[r] * corr + ps;
            mrow[r] = mnew;
#pragma unroll
            for (int nt = 0; nt < 4; ++nt)
                oacc[nt][r] = oacc[nt][r] * corr;

            // stage P (C-layout element -> row-major LDS)
            Pst[(r + 8 * lh) * LDS_STRIDE + ln]      = (__bf16)e0;
            Pst[(r + 8 * lh) * LDS_STRIDE + 16 + ln] = (__bf16)e1;
        }
        __syncthreads();

        // ---- P fragment (A-layout) from LDS ----
        FragU fp;
        fp.q[0] = *(const uint4*)(&Pst[ln * LDS_STRIDE + lh * 8]);
        fp.q[1] = *(const uint4*)(&Pst[ln * LDS_STRIDE + 16 + lh * 8]);
        const v16bf ap = fp.v;

        // ---- V fragments straight from global (pre-transposed [dk][S]) ----
#pragma unroll
        for (int nt = 0; nt < 4; ++nt) {
            const __bf16* vrow = vbT + (size_t)(nt * 16 + ln) * S + kv;
            FragU fv;
            fv.q[0] = *(const uint4*)(vrow + lh * 8);
            fv.q[1] = *(const uint4*)(vrow + 16 + lh * 8);
            oacc[nt] = WMMA_BF16(ap, fv.v, oacc[nt]);
        }
        __syncthreads();
        buf ^= 1;
    }

    // ---- epilogue: normalize and store to attn[B,S,H*dk] (bf16) ----
#pragma unroll
    for (int nt = 0; nt < 4; ++nt) {
#pragma unroll
        for (int r = 0; r < 8; ++r) {
            const int srow = q0 + r + 8 * lh;
            const float val = oacc[nt][r] / lrow[r];
            attn[(((size_t)b * S + srow) * H + h) * DK + nt * 16 + ln] = (__bf16)val;
        }
    }
}

// ---------------------------------------------------------------------------
// Launch: 3x (GEMM + pack), flash attention, output GEMM.
// Workspace (bf16, 16 MiB each, 80 MiB total):
//   qp [B,H,S,dk] | kp [B,H,S,dk] | vp [B,H,dk,S] | tmp [B,S,D] | attn [B,S,D]
// ---------------------------------------------------------------------------
extern "C" void kernel_launch(void* const* d_in, const int* in_sizes, int n_in,
                              void* d_out, int out_size, void* d_ws, size_t ws_size,
                              hipStream_t stream)
{
    const float* x  = (const float*)d_in[0];
    const float* wq = (const float*)d_in[1];
    const float* wk = (const float*)d_in[2];
    const float* wv = (const float*)d_in[3];
    const float* wo = (const float*)d_in[4];
    float* out      = (float*)d_out;

    char* ws = (char*)d_ws;
    const size_t SZ = (size_t)Bsz * H * S * DK * 2; // 16 MiB per bf16 tensor
    unsigned short* qp   = (unsigned short*)(ws + 0 * SZ);
    unsigned short* kp   = (unsigned short*)(ws + 1 * SZ);
    unsigned short* vp   = (unsigned short*)(ws + 2 * SZ);
    unsigned short* tmp  = (unsigned short*)(ws + 3 * SZ);
    unsigned short* attn = (unsigned short*)(ws + 4 * SZ);

    const dim3 gemm_grid(64, 8, 1);   // M/128 x N/128
    const dim3 gemm_blk(256, 1, 1);
    const dim3 pack_grid(Bsz * S, 1, 1);
    const dim3 pack_blk(256, 1, 1);
    const dim3 fa_grid(Bsz * H * (S / 16), 1, 1);
    const dim3 fa_blk(32, 1, 1);

    // Q projection + RoPE pack
    gemm_wmma<false, false><<<gemm_grid, gemm_blk, 0, stream>>>(x, wq, tmp);
    rope_pack<<<pack_grid, pack_blk, 0, stream>>>(tmp, qp, 1, 0);
    // K projection + RoPE pack
    gemm_wmma<false, false><<<gemm_grid, gemm_blk, 0, stream>>>(x, wk, tmp);
    rope_pack<<<pack_grid, pack_blk, 0, stream>>>(tmp, kp, 1, 0);
    // V projection + transposed pack (no rope)
    gemm_wmma<false, false><<<gemm_grid, gemm_blk, 0, stream>>>(x, wv, tmp);
    rope_pack<<<pack_grid, pack_blk, 0, stream>>>(tmp, vp, 0, 1);
    // Attention
    flash_attn<<<fa_grid, fa_blk, 0, stream>>>(qp, kp, vp, attn);
    // Output projection (TDM-staged bf16 A, f32 out)
    gemm_wmma<true, true><<<gemm_grid, gemm_blk, 0, stream>>>(attn, wo, out);
}
